// MLPGraphNetwork_2010044695200
// MI455X (gfx1250) — matrix operand order repaired and checked by reading, not compiled, not run on hardware
//
#include <hip/hip_runtime.h>

#define N_NODES 100000
#define N_EDGES 800000
#define FEAT 64

typedef __attribute__((ext_vector_type(16))) __bf16 v16bf;
typedef __attribute__((ext_vector_type(8)))  float  v8f;
typedef __attribute__((ext_vector_type(4)))  unsigned int v4u;

union Frag { v16bf bf; v4u u[2]; };

__device__ __forceinline__ unsigned short f2bf(float a) {
    unsigned u = __float_as_uint(a);
    return (unsigned short)((u + 0x7FFFu + ((u >> 16) & 1u)) >> 16);
}
__device__ __forceinline__ unsigned f2bf_pk(float a, float b) {
    return (unsigned)f2bf(a) | ((unsigned)f2bf(b) << 16);
}
__device__ __forceinline__ void wave_lds_fence() {
    __asm__ volatile("s_wait_dscnt 0" ::: "memory");
}
// Guaranteed-native global fp32 atomic add (no-return form, STOREcnt-tracked).
__device__ __forceinline__ void gatomic_fadd(float* p, float v) {
    __asm__ volatile("global_atomic_add_f32 %0, %1, off" :: "v"(p), "v"(v) : "memory");
}
#define WMMA_BF16(A, B, C) \
    __builtin_amdgcn_wmma_f32_16x16x32_bf16(false, (A), false, (B), (short)0, (C), false, false)

// ---------------------------------------------------------------------------
// Pack fp32 row-major W[K][64] into bf16 WMMA B-fragments:
// frag(kt,nt): lane l holds column nt*16+(l&15), K = kt*32 + (l>>4)*8 + (j<4 ? 2j : 16+2(j-4)) .. +1
// dst[t] with t = kt*1024 + nt*256 + lane*8 + j
// ---------------------------------------------------------------------------
__global__ void pack_wb(const float* __restrict__ W, int ktCount, unsigned* __restrict__ dst) {
    int t = blockIdx.x * blockDim.x + threadIdx.x;
    if (t >= ktCount * 1024) return;
    int j = t & 7, lane = (t >> 3) & 31, nt = (t >> 8) & 3, kt = t >> 10;
    int n  = nt * 16 + (lane & 15);
    int k0 = kt * 32 + ((lane >> 4) << 3) + (j < 4 ? 2 * j : 16 + 2 * (j - 4));
    dst[t] = f2bf_pk(W[(size_t)k0 * 64 + n], W[(size_t)(k0 + 1) * 64 + n]);
}

__global__ void zero_kernel(float* __restrict__ agg, long n, float* __restrict__ sums) {
    long i = (long)blockIdx.x * blockDim.x + threadIdx.x;
    if (i < n) agg[i] = 0.0f;
    if (i < 128) sums[i] = 0.0f;
}

// ---------------------------------------------------------------------------
// Edge MLP: one wave per 16-edge tile, 4 waves/block.
// in  = [edges | nodes[recv] | nodes[send] | g]  (256 feats)
// out = relu(in@We1+be1)@We2+be2
// Also: atomic scatter-add into agg[recv], fused column-sum into e_sum.
// ---------------------------------------------------------------------------
__launch_bounds__(128)
__global__ void edge_mlp_kernel(const float* __restrict__ edges,
                                const float* __restrict__ nodes,
                                const float* __restrict__ g,
                                const int* __restrict__ senders,
                                const int* __restrict__ receivers,
                                const unsigned* __restrict__ wsWe1,
                                const unsigned* __restrict__ wsWe2,
                                const float* __restrict__ be1,
                                const float* __restrict__ be2,
                                float* __restrict__ out_edges,
                                float* __restrict__ agg,
                                float* __restrict__ e_sum) {
    __shared__ unsigned short sIn[4][16][264];  // 256 + 8 pad
    __shared__ unsigned short sH[4][16][72];    // 64 + 8 pad
    __shared__ float sRed[4][32][4];            // per-lane column partials
    const int lane = threadIdx.x & 31;
    const int w    = threadIdx.x >> 5;
    const int tile = blockIdx.x * 4 + w;
    const int e0   = tile * 16;
    const int half = lane >> 4;
    const int lmod = lane & 15;

    // ---- stage 16x256 fp32 -> bf16 LDS ----
#pragma unroll
    for (int i = 0; i < 32; ++i) {
        int row   = i >> 1;
        int col4  = (i & 1) * 32 + lane;  // 0..63 float4 slots across 256 feats
        int seg   = col4 >> 4;
        int idx16 = col4 & 15;
        int e     = e0 + row;
        const float* src;
        if (seg == 0)      src = edges + (size_t)e * FEAT;
        else if (seg == 1) src = nodes + (size_t)receivers[e] * FEAT;
        else if (seg == 2) src = nodes + (size_t)senders[e] * FEAT;
        else               src = g;
        const float4 v = *(const float4*)(src + idx16 * 4);
        unsigned* dst = (unsigned*)&sIn[w][row][col4 * 4];
        dst[0] = f2bf_pk(v.x, v.y);
        dst[1] = f2bf_pk(v.z, v.w);
    }
    wave_lds_fence();

    // ---- layer 1: 256 -> 64 ----
    v8f acc[4];
#pragma unroll
    for (int nt = 0; nt < 4; ++nt) {
        float b = be1[nt * 16 + lmod];
#pragma unroll
        for (int r = 0; r < 8; ++r) acc[nt][r] = b;
    }
#pragma unroll
    for (int kt = 0; kt < 8; ++kt) {
        Frag A;
        const unsigned short* rowp = &sIn[w][lmod][0];
        int k0 = kt * 32 + half * 8;
        A.u[0] = *(const v4u*)(rowp + k0);
        A.u[1] = *(const v4u*)(rowp + k0 + 16);
#pragma unroll
        for (int nt = 0; nt < 4; ++nt) {
            Frag B;
            const v4u* wp = (const v4u*)(wsWe1 + ((size_t)((kt * 4 + nt) * 32 + lane)) * 8);
            B.u[0] = wp[0]; B.u[1] = wp[1];
            acc[nt] = WMMA_BF16(A.bf, B.bf, acc[nt]);
        }
    }
    // relu -> bf16 h in LDS
#pragma unroll
    for (int nt = 0; nt < 4; ++nt) {
        int n = nt * 16 + lmod;
#pragma unroll
        for (int r = 0; r < 8; ++r) {
            float v = acc[nt][r];
            v = v > 0.0f ? v : 0.0f;
            sH[w][r + half * 8][n] = f2bf(v);
        }
    }
    wave_lds_fence();

    // ---- layer 2: 64 -> 64 ----
    v8f acc2[4];
#pragma unroll
    for (int nt = 0; nt < 4; ++nt) {
        float b = be2[nt * 16 + lmod];
#pragma unroll
        for (int r = 0; r < 8; ++r) acc2[nt][r] = b;
    }
#pragma unroll
    for (int kt = 0; kt < 2; ++kt) {
        Frag A;
        const unsigned short* rowp = &sH[w][lmod][0];
        int k0 = kt * 32 + half * 8;
        A.u[0] = *(const v4u*)(rowp + k0);
        A.u[1] = *(const v4u*)(rowp + k0 + 16);
#pragma unroll
        for (int nt = 0; nt < 4; ++nt) {
            Frag B;
            const v4u* wp = (const v4u*)(wsWe2 + ((size_t)((kt * 4 + nt) * 32 + lane)) * 8);
            B.u[0] = wp[0]; B.u[1] = wp[1];
            acc2[nt] = WMMA_BF16(A.bf, B.bf, acc2[nt]);
        }
    }

    // ---- store new_edges + scatter-add into agg[recv] + column partials ----
#pragma unroll
    for (int nt = 0; nt < 4; ++nt) {
        int n = nt * 16 + lmod;
        float ps = 0.0f;
#pragma unroll
        for (int r = 0; r < 8; ++r) {
            int m   = r + half * 8;
            float v = acc2[nt][r];
            ps += v;
            out_edges[(size_t)(e0 + m) * FEAT + n] = v;
            int ridx = receivers[e0 + m];
            gatomic_fadd(&agg[(size_t)ridx * FEAT + n], v);
        }
        sRed[w][lane][nt] = ps;
    }
    __syncthreads();
    // block-level column sum -> 64 global atomics per block
    if (threadIdx.x < 64) {
        int col = threadIdx.x;
        int nt = col >> 4, li = col & 15;
        float t = 0.0f;
#pragma unroll
        for (int ww = 0; ww < 4; ++ww)
            t += sRed[ww][li][nt] + sRed[ww][li + 16][nt];
        gatomic_fadd(&e_sum[col], t);
    }
}

// ---------------------------------------------------------------------------
// Node MLP: in = [agg | nodes | g] (192 feats), writes new_nodes in place
// over the agg buffer (each wave reads only its own 16 rows first).
// ---------------------------------------------------------------------------
__launch_bounds__(128)
__global__ void node_mlp_kernel(const float* __restrict__ nodes,
                                const float* __restrict__ g,
                                const unsigned* __restrict__ wsWn1,
                                const unsigned* __restrict__ wsWn2,
                                const float* __restrict__ bn1,
                                const float* __restrict__ bn2,
                                float* __restrict__ agg_out) {
    __shared__ unsigned short sIn[4][16][200];  // 192 + 8 pad
    __shared__ unsigned short sH[4][16][72];
    const int lane = threadIdx.x & 31;
    const int w    = threadIdx.x >> 5;
    const int tile = blockIdx.x * 4 + w;
    if (tile >= N_NODES / 16) return;
    const int v0   = tile * 16;
    const int half = lane >> 4;
    const int lmod = lane & 15;

#pragma unroll
    for (int i = 0; i < 24; ++i) {
        int slot  = i * 32 + lane;  // 0..767 = 16 rows * 48 float4 slots
        int row   = slot / 48;
        int col4  = slot % 48;
        int seg   = col4 >> 4;
        int idx16 = col4 & 15;
        const float* src;
        if (seg == 0)      src = agg_out + (size_t)(v0 + row) * FEAT;
        else if (seg == 1) src = nodes + (size_t)(v0 + row) * FEAT;
        else               src = g;
        const float4 v = *(const float4*)(src + idx16 * 4);
        unsigned* dst = (unsigned*)&sIn[w][row][col4 * 4];
        dst[0] = f2bf_pk(v.x, v.y);
        dst[1] = f2bf_pk(v.z, v.w);
    }
    wave_lds_fence();

    v8f acc[4];
#pragma unroll
    for (int nt = 0; nt < 4; ++nt) {
        float b = bn1[nt * 16 + lmod];
#pragma unroll
        for (int r = 0; r < 8; ++r) acc[nt][r] = b;
    }
#pragma unroll
    for (int kt = 0; kt < 6; ++kt) {
        Frag A;
        const unsigned short* rowp = &sIn[w][lmod][0];
        int k0 = kt * 32 + half * 8;
        A.u[0] = *(const v4u*)(rowp + k0);
        A.u[1] = *(const v4u*)(rowp + k0 + 16);
#pragma unroll
        for (int nt = 0; nt < 4; ++nt) {
            Frag B;
            const v4u* wp = (const v4u*)(wsWn1 + ((size_t)((kt * 4 + nt) * 32 + lane)) * 8);
            B.u[0] = wp[0]; B.u[1] = wp[1];
            acc[nt] = WMMA_BF16(A.bf, B.bf, acc[nt]);
        }
    }
#pragma unroll
    for (int nt = 0; nt < 4; ++nt) {
        int n = nt * 16 + lmod;
#pragma unroll
        for (int r = 0; r < 8; ++r) {
            float v = acc[nt][r];
            v = v > 0.0f ? v : 0.0f;
            sH[w][r + half * 8][n] = f2bf(v);
        }
    }
    wave_lds_fence();

    v8f acc2[4];
#pragma unroll
    for (int nt = 0; nt < 4; ++nt) {
        float b = bn2[nt * 16 + lmod];
#pragma unroll
        for (int r = 0; r < 8; ++r) acc2[nt][r] = b;
    }
#pragma unroll
    for (int kt = 0; kt < 2; ++kt) {
        Frag A;
        const unsigned short* rowp = &sH[w][lmod][0];
        int k0 = kt * 32 + half * 8;
        A.u[0] = *(const v4u*)(rowp + k0);
        A.u[1] = *(const v4u*)(rowp + k0 + 16);
#pragma unroll
        for (int nt = 0; nt < 4; ++nt) {
            Frag B;
            const v4u* wp = (const v4u*)(wsWn2 + ((size_t)((kt * 4 + nt) * 32 + lane)) * 8);
            B.u[0] = wp[0]; B.u[1] = wp[1];
            acc2[nt] = WMMA_BF16(A.bf, B.bf, acc2[nt]);
        }
    }
#pragma unroll
    for (int nt = 0; nt < 4; ++nt) {
        int n = nt * 16 + lmod;
#pragma unroll
        for (int r = 0; r < 8; ++r) {
            int m = r + half * 8;
            agg_out[(size_t)(v0 + m) * FEAT + n] = acc2[nt][r];
        }
    }
}

// ---------------------------------------------------------------------------
// Column sums of X[rows][64] into sum[64] (sum pre-zeroed).
// ---------------------------------------------------------------------------
__global__ void reduce_cols(const float* __restrict__ X, long rows, float* __restrict__ sum) {
    __shared__ float s[256];
    int col = threadIdx.x & 63;
    int rg  = threadIdx.x >> 6;  // 0..3
    float acc = 0.0f;
    long stride = (long)gridDim.x * 4;
    for (long r = (long)blockIdx.x * 4 + rg; r < rows; r += stride)
        acc += X[r * 64 + col];
    s[threadIdx.x] = acc;
    __syncthreads();
    if (threadIdx.x < 64) {
        float t = s[threadIdx.x] + s[threadIdx.x + 64] + s[threadIdx.x + 128] + s[threadIdx.x + 192];
        gatomic_fadd(&sum[col], t);
    }
}

// ---------------------------------------------------------------------------
// Global MLP: x = [e_mean | n_mean | g] (192) -> 64 -> relu -> 64
// ---------------------------------------------------------------------------
__global__ void global_mlp_kernel(const float* __restrict__ e_sum,
                                  const float* __restrict__ n_sum,
                                  const float* __restrict__ g,
                                  const float* __restrict__ Wg1, const float* __restrict__ bg1,
                                  const float* __restrict__ Wg2, const float* __restrict__ bg2,
                                  float* __restrict__ out) {
    __shared__ float x[192];
    __shared__ float h[64];
    int t = threadIdx.x;
    x[t]       = e_sum[t] * (1.0f / (float)N_EDGES);
    x[64 + t]  = n_sum[t] * (1.0f / (float)N_NODES);
    x[128 + t] = g[t];
    __syncthreads();
    float a = bg1[t];
    for (int k = 0; k < 192; ++k) a = fmaf(x[k], Wg1[k * 64 + t], a);
    h[t] = a > 0.0f ? a : 0.0f;
    __syncthreads();
    float o = bg2[t];
    for (int k = 0; k < 64; ++k) o = fmaf(h[k], Wg2[k * 64 + t], o);
    out[t] = o;
}

extern "C" void kernel_launch(void* const* d_in, const int* in_sizes, int n_in,
                              void* d_out, int out_size, void* d_ws, size_t ws_size,
                              hipStream_t stream) {
    const float* nodes     = (const float*)d_in[0];
    const float* edges     = (const float*)d_in[1];
    const float* g         = (const float*)d_in[2];
    const int*   senders   = (const int*)d_in[3];
    const int*   receivers = (const int*)d_in[4];
    const float* We1 = (const float*)d_in[5];  const float* be1 = (const float*)d_in[6];
    const float* We2 = (const float*)d_in[7];  const float* be2 = (const float*)d_in[8];
    const float* Wn1 = (const float*)d_in[9];  const float* bn1 = (const float*)d_in[10];
    const float* Wn2 = (const float*)d_in[11]; const float* bn2 = (const float*)d_in[12];
    const float* Wg1 = (const float*)d_in[13]; const float* bg1 = (const float*)d_in[14];
    const float* Wg2 = (const float*)d_in[15]; const float* bg2 = (const float*)d_in[16];

    float* out       = (float*)d_out;
    float* out_edges = out;                                   // [800000, 64]
    float* out_nodes = out + (size_t)N_EDGES * FEAT;          // [100000, 64] (also agg buffer)
    float* out_glob  = out_nodes + (size_t)N_NODES * FEAT;    // [64]

    unsigned* ws     = (unsigned*)d_ws;
    unsigned* wsWe1  = ws;           // 8*1024 dwords
    unsigned* wsWe2  = ws + 8192;    // 2*1024
    unsigned* wsWn1  = ws + 10240;   // 6*1024
    unsigned* wsWn2  = ws + 16384;   // 2*1024
    float*    e_sum  = (float*)(ws + 18432);  // 64
    float*    n_sum  = (float*)(ws + 18496);  // 64

    pack_wb<<<(8 * 1024 + 255) / 256, 256, 0, stream>>>(We1, 8, wsWe1);
    pack_wb<<<(2 * 1024 + 255) / 256, 256, 0, stream>>>(We2, 2, wsWe2);
    pack_wb<<<(6 * 1024 + 255) / 256, 256, 0, stream>>>(Wn1, 6, wsWn1);
    pack_wb<<<(2 * 1024 + 255) / 256, 256, 0, stream>>>(Wn2, 2, wsWn2);

    zero_kernel<<<((long)N_NODES * FEAT + 255) / 256, 256, 0, stream>>>(
        out_nodes, (long)N_NODES * FEAT, e_sum);  // zeros e_sum..n_sum (128 floats)

    edge_mlp_kernel<<<N_EDGES / 16 / 4, 128, 0, stream>>>(
        edges, nodes, g, senders, receivers, wsWe1, wsWe2, be1, be2,
        out_edges, out_nodes, e_sum);

    node_mlp_kernel<<<(N_NODES / 16 + 3) / 4, 128, 0, stream>>>(
        nodes, g, wsWn1, wsWn2, bn1, bn2, out_nodes);

    reduce_cols<<<256, 256, 0, stream>>>(out_nodes, (long)N_NODES, n_sum);

    global_mlp_kernel<<<1, 64, 0, stream>>>(e_sum, n_sum, g, Wg1, bg1, Wg2, bg2, out_glob);
}
